// MultiHeadAttention_46712064311658
// MI455X (gfx1250) — compile-verified, matrix operand order checked by reading, not run on hardware
//
#include <hip/hip_runtime.h>
#include <hip/hip_bf16.h>

#define T_DIM 4096
#define E_DIM 768
#define H_DIM 12
#define DH_DIM 64

typedef __attribute__((ext_vector_type(16))) _Float16 v16h;
typedef __attribute__((ext_vector_type(8)))  _Float16 v8h;
typedef __attribute__((ext_vector_type(8)))  float    v8f;
typedef __attribute__((ext_vector_type(4)))  int      v4i;

#if __has_builtin(__builtin_amdgcn_global_load_async_to_lds_b128) && \
    __has_builtin(__builtin_amdgcn_s_wait_asynccnt)
#define USE_ASYNC_LDS 1
#endif

#ifdef USE_ASYNC_LDS
#define WAIT_ASYNC(n) __builtin_amdgcn_s_wait_asynccnt(n)
#else
#define WAIT_ASYNC(n)
#endif

// Copy 16 bytes global -> LDS (async DMA when available, tracked by ASYNCcnt).
// The builtin expects int4-typed pointers; clang rewrites generic->global/LDS
// address spaces automatically for builtin pointer params.
__device__ __forceinline__ void copy_b128(const _Float16* g, _Float16* l) {
#ifdef USE_ASYNC_LDS
  __builtin_amdgcn_global_load_async_to_lds_b128((v4i*)g, (v4i*)l, 0, 0);
#else
  *(v8h*)l = *(const v8h*)g;
#endif
}

// D = A(16x32) * B(32x16) + C, f16 inputs, f32 accumulate
__device__ __forceinline__ v8f wmma_f16(v16h a, v16h b, v8f c) {
  return __builtin_amdgcn_wmma_f32_16x16x32_f16(
      false, a, false, b, (short)0, c, false, false);
}

// Load a 16x32 fragment (rows x K) from a row-major matrix with leading dim ld.
// Works for A (rows = M) and B (rows = N; column n of the K-major B operand is a
// contiguous row in memory). 16-bit layout: lanes 0-15 hold row (lane&15),
// K = 0..7 (v0-3) and 16..23 (v4-7); lanes 16-31 hold K = 8..15 and 24..31.
__device__ __forceinline__ v16h load_frag(const _Float16* p, int ld) {
  const int lane = threadIdx.x & 31;
  const int row  = lane & 15;
  const int k0   = (lane >> 4) * 8;
  const _Float16* r = p + row * ld + k0;
  v8h lo = *(const v8h*)(r);
  v8h hi = *(const v8h*)(r + 16);
  v16h out;
#pragma unroll
  for (int i = 0; i < 8; ++i) { out[i] = lo[i]; out[i + 8] = hi[i]; }
  return out;
}

// Stage a 64-row x 32-col f16 tile (row-major, ld=ldg) into LDS (ld=32).
// 128 threads, 32 bytes each, 2 async b128 per thread -> 2 instr per wave.
__device__ __forceinline__ void stage_64x32(const _Float16* g, int ldg, _Float16* l) {
  const int tid = threadIdx.x & 127;
  const int row = tid >> 1;
  const int col = (tid & 1) * 16;
  const _Float16* gp = g + (size_t)row * ldg + col;
  _Float16* lp = l + row * 32 + col;
  copy_b128(gp, lp);
  copy_b128(gp + 8, lp + 8);
}

// Stage a 32-row x 64-col f16 tile (row-major, ld=ldg) into LDS (ld=64).
__device__ __forceinline__ void stage_32x64(const _Float16* g, int ldg, _Float16* l) {
  const int tid = threadIdx.x & 127;
  const int row = tid >> 2;
  const int col = (tid & 3) * 16;
  const _Float16* gp = g + (size_t)row * ldg + col;
  _Float16* lp = l + row * 64 + col;
  copy_b128(gp, lp);
  copy_b128(gp + 8, lp + 8);
}

// 64x64 output tile, 4 waves: wave w computes rows [16w,16w+16) x 64 cols.
// A/B k-slices double-buffered in LDS via async DMA.
__device__ __forceinline__ void gemm_acc_64x64(const _Float16* Ag, const _Float16* Wg,
                                               int K,
                                               _Float16 (*ldsA)[64 * 32],
                                               _Float16 (*ldsB)[64 * 32],
                                               v8f* acc) {
  const int wave = (threadIdx.x >> 5) & 3;
  const int NK = K >> 5;
  stage_64x32(Ag, K, ldsA[0]);
  stage_64x32(Wg, K, ldsB[0]);
  for (int kt = 0; kt < NK; ++kt) {
    const int cur = kt & 1;
    if (kt + 1 < NK) {
      stage_64x32(Ag + (kt + 1) * 32, K, ldsA[cur ^ 1]);
      stage_64x32(Wg + (kt + 1) * 32, K, ldsB[cur ^ 1]);
      WAIT_ASYNC(4);          // current tile done; next tile still in flight
    } else {
      WAIT_ASYNC(0);
    }
    __syncthreads();
    v16h a = load_frag(ldsA[cur] + wave * 16 * 32, 32);
#pragma unroll
    for (int j = 0; j < 4; ++j) {
      v16h b = load_frag(ldsB[cur] + j * 16 * 32, 32);
      acc[j] = wmma_f16(a, b, acc[j]);
    }
    __syncthreads();          // all reads done before buffer is re-filled
  }
}

// ---------------------------------------------------------------- convert ----
__global__ void cvt_f32_f16_kernel(const float* __restrict__ src,
                                   _Float16* __restrict__ dst, int n) {
  int i = blockIdx.x * blockDim.x + threadIdx.x;
  if (i < n) dst[i] = (_Float16)src[i];
}

// ------------------------------------------------------------- QKV GEMMs ----
// y = x @ W^T + b (M=4096,N=768,K=768). Scattered per-head outputs:
//   mat 0: qh[h][t][d]   mat 1: kh[h][t][d]   mat 2: vth[h][d][t]
__global__ __launch_bounds__(128)
void qkv_proj_kernel(const _Float16* __restrict__ xq,
                     const _Float16* __restrict__ xv,
                     const _Float16* __restrict__ w16,
                     const float* __restrict__ bq,
                     const float* __restrict__ bk,
                     const float* __restrict__ bv,
                     _Float16* __restrict__ qh,
                     _Float16* __restrict__ kh,
                     _Float16* __restrict__ vth) {
  __shared__ __align__(16) _Float16 ldsA[2][64 * 32];
  __shared__ __align__(16) _Float16 ldsB[2][64 * 32];

  const int MT = T_DIM / 64;   // 64
  const int NT = E_DIM / 64;   // 12
  int bid = blockIdx.x;
  int mat = bid / (MT * NT);
  int rem = bid % (MT * NT);
  int m0 = (rem / NT) * 64;
  int n0 = (rem % NT) * 64;

  const _Float16* A = (mat == 0) ? xq : xv;
  const _Float16* W = w16 + (size_t)mat * E_DIM * E_DIM;
  const float* bias = (mat == 0) ? bq : ((mat == 1) ? bk : bv);

  v8f acc[4] = {{}, {}, {}, {}};
  gemm_acc_64x64(A + (size_t)m0 * E_DIM, W + (size_t)n0 * E_DIM, E_DIM,
                 ldsA, ldsB, acc);

  const int wave = threadIdx.x >> 5;
  const int lane = threadIdx.x & 31;
  const int half = lane >> 4;
  const int col  = lane & 15;
#pragma unroll
  for (int j = 0; j < 4; ++j) {
    int e = n0 + j * 16 + col;
    float bb = bias[e];
    int h = e >> 6, d = e & 63;
#pragma unroll
    for (int r = 0; r < 8; ++r) {
      int m = m0 + wave * 16 + r + 8 * half;
      float y = acc[j][r] + bb;
      if (mat == 0)
        qh[(size_t)h * T_DIM * DH_DIM + (size_t)m * DH_DIM + d] = (_Float16)y;
      else if (mat == 1)
        kh[(size_t)h * T_DIM * DH_DIM + (size_t)m * DH_DIM + d] = (_Float16)y;
      else
        vth[(size_t)h * DH_DIM * T_DIM + (size_t)d * T_DIM + m] = (_Float16)y;
    }
  }
}

// ------------------------------------------------------ flash attention -----
// 4 waves per block, 64 queries per block (wave w: rows q0blk+16w..+15).
// K (32x64) and V^T (64x32) tiles double-buffered in LDS, shared by all waves.
__global__ __launch_bounds__(128)
void attn_kernel(const _Float16* __restrict__ qh,
                 const _Float16* __restrict__ kh,
                 const _Float16* __restrict__ vth,
                 _Float16* __restrict__ ctx16) {
  __shared__ __align__(16) _Float16 kbuf[2][32 * 64];  // keys x d   (ld 64)
  __shared__ __align__(16) _Float16 vbuf[2][64 * 32];  // d    x key (ld 32)
  __shared__ __align__(16) _Float16 pbuf[4][16 * 32];  // per-wave P (ld 32)

  const int QT = T_DIM / 64;               // 64 query blocks per head
  int h     = blockIdx.x / QT;
  int qb    = blockIdx.x % QT;
  int q0blk = qb * 64;
  const int wave = threadIdx.x >> 5;
  const int q0   = q0blk + wave * 16;      // this wave's query rows

  const _Float16* Q  = qh  + (size_t)h * T_DIM * DH_DIM;
  const _Float16* Kp = kh  + (size_t)h * T_DIM * DH_DIM;
  const _Float16* Vt = vth + (size_t)h * DH_DIM * T_DIM;

  const int lane = threadIdx.x & 31;
  const int half = lane >> 4;
  const int col  = lane & 15;

  v16h aq0 = load_frag(Q + (size_t)q0 * DH_DIM +  0, DH_DIM);
  v16h aq1 = load_frag(Q + (size_t)q0 * DH_DIM + 32, DH_DIM);

  v8f o0 = {}, o1 = {}, o2 = {}, o3 = {};
  float rm[8], rs[8];
#pragma unroll
  for (int r = 0; r < 8; ++r) { rm[r] = -3.0e38f; rs[r] = 0.0f; }

  const int nblk = qb * 2 + 2;             // key blocks of 32 covering diagonal
  stage_32x64(Kp, DH_DIM, kbuf[0]);        // keys 0..31
  stage_64x32(Vt, T_DIM, vbuf[0]);

  for (int jt = 0; jt < nblk; ++jt) {
    const int j0 = jt * 32;
    const int cur = jt & 1;
    if (jt + 1 < nblk) {
      stage_32x64(Kp + (size_t)(j0 + 32) * DH_DIM, DH_DIM, kbuf[cur ^ 1]);
      stage_64x32(Vt + (j0 + 32), T_DIM, vbuf[cur ^ 1]);
      WAIT_ASYNC(4);
    } else {
      WAIT_ASYNC(0);
    }
    __syncthreads();

    if (j0 <= q0 + 15) {                   // wave-uniform causal skip
      // ---- S = Q @ K^T from LDS K tile
      v8f s0 = {}, s1 = {};
      s0 = wmma_f16(aq0, load_frag(kbuf[cur] +  0 * 64 +  0, 64), s0);
      s0 = wmma_f16(aq1, load_frag(kbuf[cur] +  0 * 64 + 32, 64), s0);
      s1 = wmma_f16(aq0, load_frag(kbuf[cur] + 16 * 64 +  0, 64), s1);
      s1 = wmma_f16(aq1, load_frag(kbuf[cur] + 16 * 64 + 32, 64), s1);

      const bool domask = (j0 + 31 > q0);
#pragma unroll
      for (int r = 0; r < 8; ++r) {
        int m = q0 + r + 8 * half;
        float v0 = s0[r] * 0.125f;         // 1/sqrt(64)
        float v1 = s1[r] * 0.125f;
        if (domask) {
          if (j0 + col      > m) v0 = -1.0e9f;
          if (j0 + 16 + col > m) v1 = -1.0e9f;
        }
        float mx = fmaxf(v0, v1);
        mx = fmaxf(mx, __shfl_xor(mx, 1, 32));
        mx = fmaxf(mx, __shfl_xor(mx, 2, 32));
        mx = fmaxf(mx, __shfl_xor(mx, 4, 32));
        mx = fmaxf(mx, __shfl_xor(mx, 8, 32));
        float nm = fmaxf(rm[r], mx);
        float al = __expf(rm[r] - nm);
        float p0 = __expf(v0 - nm);
        float p1 = __expf(v1 - nm);
        float sm = p0 + p1;
        sm += __shfl_xor(sm, 1, 32);
        sm += __shfl_xor(sm, 2, 32);
        sm += __shfl_xor(sm, 4, 32);
        sm += __shfl_xor(sm, 8, 32);
        rs[r] = rs[r] * al + sm;
        rm[r] = nm;
        o0[r] *= al; o1[r] *= al; o2[r] *= al; o3[r] *= al;
        int ml = r + 8 * half;
        pbuf[wave][ml * 32 + col]      = (_Float16)p0;
        pbuf[wave][ml * 32 + 16 + col] = (_Float16)p1;
      }

      // ---- O += P @ V from LDS (P private per wave; same-wave DS is in-order)
      v16h ap = load_frag(pbuf[wave], 32);
      o0 = wmma_f16(ap, load_frag(vbuf[cur] +  0 * 32, 32), o0);
      o1 = wmma_f16(ap, load_frag(vbuf[cur] + 16 * 32, 32), o1);
      o2 = wmma_f16(ap, load_frag(vbuf[cur] + 32 * 32, 32), o2);
      o3 = wmma_f16(ap, load_frag(vbuf[cur] + 48 * 32, 32), o3);
    }
    __syncthreads();
  }

#pragma unroll
  for (int r = 0; r < 8; ++r) {
    float inv = 1.0f / rs[r];
    int m = q0 + r + 8 * half;
    _Float16* dst = ctx16 + (size_t)m * E_DIM + h * DH_DIM;
    dst[ 0 + col] = (_Float16)(o0[r] * inv);
    dst[16 + col] = (_Float16)(o1[r] * inv);
    dst[32 + col] = (_Float16)(o2[r] * inv);
    dst[48 + col] = (_Float16)(o3[r] * inv);
  }
}

// --------------------------------------------------------- output GEMM ------
__global__ __launch_bounds__(128)
void out_proj_kernel(const _Float16* __restrict__ ctx16,
                     const _Float16* __restrict__ wo16,
                     const float* __restrict__ bo,
                     float* __restrict__ out) {
  __shared__ __align__(16) _Float16 ldsA[2][64 * 32];
  __shared__ __align__(16) _Float16 ldsB[2][64 * 32];

  const int NT = E_DIM / 64;
  int m0 = (blockIdx.x / NT) * 64;
  int n0 = (blockIdx.x % NT) * 64;

  v8f acc[4] = {{}, {}, {}, {}};
  gemm_acc_64x64(ctx16 + (size_t)m0 * E_DIM, wo16 + (size_t)n0 * E_DIM, E_DIM,
                 ldsA, ldsB, acc);

  const int wave = threadIdx.x >> 5;
  const int lane = threadIdx.x & 31;
  const int half = lane >> 4;
  const int col  = lane & 15;
#pragma unroll
  for (int j = 0; j < 4; ++j) {
    int e = n0 + j * 16 + col;
    float bb = bo[e];
#pragma unroll
    for (int r = 0; r < 8; ++r) {
      int m = m0 + wave * 16 + r + 8 * half;
      out[(size_t)m * E_DIM + e] = acc[j][r] + bb;
    }
  }
}

// ---------------------------------------------------------------- launch ----
extern "C" void kernel_launch(void* const* d_in, const int* in_sizes, int n_in,
                              void* d_out, int out_size, void* d_ws, size_t ws_size,
                              hipStream_t stream) {
  (void)in_sizes; (void)n_in; (void)out_size; (void)ws_size;

  const float* queries = (const float*)d_in[0];
  const float* values  = (const float*)d_in[1];
  // d_in[2] = mask (causality computed analytically)
  const float* Wq = (const float*)d_in[3];
  const float* bq = (const float*)d_in[4];
  const float* Wk = (const float*)d_in[5];
  const float* bk = (const float*)d_in[6];
  const float* Wv = (const float*)d_in[7];
  const float* bv = (const float*)d_in[8];
  const float* Wo = (const float*)d_in[9];
  const float* bo = (const float*)d_in[10];
  float* out = (float*)d_out;

  const size_t TE = (size_t)T_DIM * E_DIM;
  const size_t EE = (size_t)E_DIM * E_DIM;

  _Float16* ws    = (_Float16*)d_ws;
  _Float16* xq16  = ws;
  _Float16* xv16  = xq16 + TE;
  _Float16* w16   = xv16 + TE;         // Wq,Wk,Wv,Wo (4*EE)
  _Float16* qh    = w16 + 4 * EE;      // [H][T][DH]
  _Float16* kh    = qh + TE;           // [H][T][DH]
  _Float16* vth   = kh + TE;           // [H][DH][T]
  _Float16* ctx16 = vth + TE;          // [T][E]

  {
    int thr = 256;
    cvt_f32_f16_kernel<<<(TE + thr - 1) / thr, thr, 0, stream>>>(queries, xq16, (int)TE);
    cvt_f32_f16_kernel<<<(TE + thr - 1) / thr, thr, 0, stream>>>(values,  xv16, (int)TE);
    cvt_f32_f16_kernel<<<(EE + thr - 1) / thr, thr, 0, stream>>>(Wq, w16 + 0 * EE, (int)EE);
    cvt_f32_f16_kernel<<<(EE + thr - 1) / thr, thr, 0, stream>>>(Wk, w16 + 1 * EE, (int)EE);
    cvt_f32_f16_kernel<<<(EE + thr - 1) / thr, thr, 0, stream>>>(Wv, w16 + 2 * EE, (int)EE);
    cvt_f32_f16_kernel<<<(EE + thr - 1) / thr, thr, 0, stream>>>(Wo, w16 + 3 * EE, (int)EE);
  }

  qkv_proj_kernel<<<3 * (T_DIM / 64) * (E_DIM / 64), 128, 0, stream>>>(
      xq16, xv16, w16, bq, bk, bv, qh, kh, vth);

  attn_kernel<<<H_DIM * (T_DIM / 64), 128, 0, stream>>>(qh, kh, vth, ctx16);

  out_proj_kernel<<<(T_DIM / 64) * (E_DIM / 64), 128, 0, stream>>>(
      ctx16, w16 + 3 * EE, bo, out);
}
